// VQVAE_37598143709220
// MI455X (gfx1250) — matrix operand order repaired and checked by reading, not compiled, not run on hardware
//
#include <hip/hip_runtime.h>
#include <math.h>

// ---------------------------------------------------------------------------
// Types for CDNA5 WMMA (wave32, 16x16x32 bf16 -> f32)
// ---------------------------------------------------------------------------
typedef __bf16 bf16_t;
typedef __attribute__((ext_vector_type(16))) __bf16 v16bf;
typedef __attribute__((ext_vector_type(8)))  float  v8f;

union FragU {
    uint4 u[2];
    v16bf v;
};

__device__ __forceinline__ float bf2f(bf16_t b) {
    unsigned short h = __builtin_bit_cast(unsigned short, b);
    unsigned u = ((unsigned)h) << 16;
    return __builtin_bit_cast(float, u);
}

__device__ __forceinline__ bf16_t f2bf(float f) {
    unsigned u = __builtin_bit_cast(unsigned, f);
    unsigned r = u + 0x7FFFu + ((u >> 16) & 1u);   // round-to-nearest-even
    unsigned short h = (unsigned short)(r >> 16);
    return __builtin_bit_cast(bf16_t, h);
}

__device__ __forceinline__ v8f wmma_bf16(const FragU& a, const FragU& b, v8f c) {
    return __builtin_amdgcn_wmma_f32_16x16x32_bf16(false, a.v, false, b.v,
                                                   (short)0, c, false, false);
}

// gfx1250 async global->LDS copy (ASYNCcnt-tracked DMA, ISA 08 section 4)
__device__ __forceinline__ void async_g2l_b128(unsigned lds_off, const void* gaddr) {
    asm volatile("global_load_async_to_lds_b128 %0, %1, off"
                 :: "v"(lds_off), "v"(gaddr) : "memory");
}
__device__ __forceinline__ void wait_asynccnt0() {
    asm volatile("s_wait_asynccnt 0x0" ::: "memory");
}

// ---------------------------------------------------------------------------
// Generic implicit-GEMM conv descriptor
//   GEMM:  D[M,N] = W[M,K] * Im2col[K,N]   (K = Cin*KH*KW, N = Bn*OHc*OWc)
//   output position: oy = y'*out_sy + out_oy (parity write for tconv)
// ---------------------------------------------------------------------------
struct ConvDesc {
    const bf16_t* X;    // (Bn, Cin, IH, IW) bf16 activations
    const bf16_t* Wm;   // [M, K] row-major bf16 weights
    const float*  bias; // [M] or nullptr
    const bf16_t* Res;  // residual, same layout as output, or nullptr
    bf16_t* Y;          // bf16 output (Bn, M, OHf, OWf) or nullptr
    float*  Yf;         // f32 output, same layout, or nullptr
    int Bn, Cin, IH, IW;
    int KH, KW, stride, pad_y, pad_x;
    int OHc, OWc;       // conv-grid output dims (OWc % 8 == 0 assumed)
    int OHf, OWf;       // full output dims
    int out_sy, out_oy, out_sx, out_ox;
    int M, K, N;
    int relu_in;        // relu applied to X on load
    int act;            // 0 none, 1 relu, 2 tanh
};

// Block tile: 128(M) x 64(N), K-step 32, double-buffered LDS with async DMA
// for the weight tile. 256 threads = 8 waves (4x2), each wave a 32x32 patch.
__global__ __launch_bounds__(256)
void conv_gemm_wmma(ConvDesc d) {
    __shared__ __align__(16) bf16_t As[2][128][40];  // [M][K], 80B row stride
    __shared__ __align__(16) bf16_t Bs[2][64][40];   // [N][K]

    const int t      = threadIdx.x;
    const int blockN = blockIdx.x * 64;
    const int blockM = blockIdx.y * 128;
    const int lane = t & 31;
    const int wv   = t >> 5;
    const int wm   = wv & 3;
    const int wn   = wv >> 2;
    const int l16  = lane & 15;
    const int lh   = lane >> 4;
    const bf16_t BZ = __builtin_bit_cast(bf16_t, (unsigned short)0);

    v8f acc[2][2] = {};

    const int khw = d.KH * d.KW;
    const int hw  = d.OHc * d.OWc;

    // ---- B staging mapping: thread = one K-row x 8 consecutive pixels ----
    const int rkl = t & 31;          // K row within tile
    const int seg = t >> 5;          // pixel octet 0..7
    const int nB0 = blockN + seg * 8;
    const bool colOK = nB0 < d.N;    // all 8 pixels share one image row
    int bb = 0, yy = 0, xx0 = 0;
    if (colOK) {
        bb = nB0 / hw;
        int rem = nB0 - bb * hw;
        yy = rem / d.OWc;
        xx0 = rem - yy * d.OWc;
    }

    auto stageA = [&](int k0, int buf) {
        for (int g = t; g < 512; g += 256) {
            int row = g >> 2;
            int kc  = (g & 3) * 8;
            int grow = blockM + row;
            int gk   = k0 + kc;
            if (grow < d.M && gk + 8 <= d.K) {
                unsigned lds = (unsigned)(size_t)&As[buf][row][kc];
                async_g2l_b128(lds, d.Wm + (size_t)grow * d.K + gk);
            } else {
                union { bf16_t b[8]; uint4 u; } tv;
                for (int j = 0; j < 8; ++j)
                    tv.b[j] = (grow < d.M && gk + j < d.K)
                                  ? d.Wm[(size_t)grow * d.K + gk + j] : BZ;
                *(uint4*)&As[buf][row][kc] = tv.u;
            }
        }
    };

    auto stageB = [&](int k0, int buf) {
        int rk = k0 + rkl;
        if (!colOK || rk >= d.K) {
            for (int j = 0; j < 8; ++j) Bs[buf][seg * 8 + j][rkl] = BZ;
            return;
        }
        int ci = rk / khw;
        int r  = rk - ci * khw;
        int ky = r / d.KW;
        int kx = r - ky * d.KW;
        int iy  = yy * d.stride + ky - d.pad_y;
        int ix0 = xx0 * d.stride + kx - d.pad_x;
        const bf16_t* src = d.X + (((size_t)bb * d.Cin + ci) * d.IH + iy) * d.IW;
        const bool rowOK = (iy >= 0) && (iy < d.IH);
        if (rowOK && d.stride == 1 && (ix0 & 7) == 0 && ix0 >= 0 && ix0 + 8 <= d.IW) {
            // vector fast path: 8 contiguous, aligned pixels in one b128
            union { uint4 u; unsigned short s[8]; } tv;
            tv.u = *(const uint4*)(src + ix0);
            if (d.relu_in) {
                for (int j = 0; j < 8; ++j)
                    if (tv.s[j] & 0x8000u) tv.s[j] = 0;
            }
            for (int j = 0; j < 8; ++j)
                Bs[buf][seg * 8 + j][rkl] = __builtin_bit_cast(bf16_t, tv.s[j]);
        } else {
            for (int j = 0; j < 8; ++j) {
                int ix = ix0 + j * d.stride;
                unsigned short hv = 0;
                if (rowOK && ix >= 0 && ix < d.IW) {
                    hv = __builtin_bit_cast(unsigned short, src[ix]);
                    if (d.relu_in && (hv & 0x8000u)) hv = 0;
                }
                Bs[buf][seg * 8 + j][rkl] = __builtin_bit_cast(bf16_t, hv);
            }
        }
    };

    // ---- software pipeline: DMA tile k+1 while WMMA consumes tile k ----
    stageA(0, 0);
    stageB(0, 0);
    wait_asynccnt0();
    __syncthreads();

    int cur = 0;
    for (int k0 = 0; k0 < d.K; k0 += 32) {
        const int nxt = cur ^ 1;
        if (k0 + 32 < d.K) {
            stageA(k0 + 32, nxt);
            stageB(k0 + 32, nxt);
        }

        FragU a0, a1, b0, b1;
        const int ar0 = wm * 32 + l16, ar1 = ar0 + 16;
        a0.u[0] = *(const uint4*)&As[cur][ar0][lh * 8];
        a0.u[1] = *(const uint4*)&As[cur][ar0][16 + lh * 8];
        a1.u[0] = *(const uint4*)&As[cur][ar1][lh * 8];
        a1.u[1] = *(const uint4*)&As[cur][ar1][16 + lh * 8];
        const int bc0 = wn * 32 + l16, bc1 = bc0 + 16;
        b0.u[0] = *(const uint4*)&Bs[cur][bc0][lh * 8];
        b0.u[1] = *(const uint4*)&Bs[cur][bc0][16 + lh * 8];
        b1.u[0] = *(const uint4*)&Bs[cur][bc1][lh * 8];
        b1.u[1] = *(const uint4*)&Bs[cur][bc1][16 + lh * 8];

        acc[0][0] = wmma_bf16(a0, b0, acc[0][0]);
        acc[0][1] = wmma_bf16(a0, b1, acc[0][1]);
        acc[1][0] = wmma_bf16(a1, b0, acc[1][0]);
        acc[1][1] = wmma_bf16(a1, b1, acc[1][1]);

        wait_asynccnt0();
        __syncthreads();
        cur = nxt;
    }

    // ---- epilogue: bias + residual + activation + store ----
    for (int tm = 0; tm < 2; ++tm)
        for (int tn = 0; tn < 2; ++tn)
            for (int r = 0; r < 8; ++r) {
                int cout = blockM + wm * 32 + tm * 16 + r + lh * 8;
                int n    = blockN + wn * 32 + tn * 16 + l16;
                if (cout >= d.M || n >= d.N) continue;
                float v = acc[tm][tn][r];
                if (d.bias) v += d.bias[cout];
                int b2  = n / hw;
                int rem = n - b2 * hw;
                int y2  = rem / d.OWc;
                int x2  = rem - y2 * d.OWc;
                int oy  = y2 * d.out_sy + d.out_oy;
                int ox  = x2 * d.out_sx + d.out_ox;
                size_t o = (((size_t)b2 * d.M + cout) * d.OHf + oy) * d.OWf + ox;
                if (d.Res) v += bf2f(d.Res[o]);
                if (d.act == 1)      v = fmaxf(v, 0.0f);
                else if (d.act == 2) v = tanhf(v);
                if (d.Y)  d.Y[o]  = f2bf(v);
                if (d.Yf) d.Yf[o] = v;
            }
}

// ---------------------------------------------------------------------------
// Helper kernels
// ---------------------------------------------------------------------------
__global__ void cvt_f32_to_bf16(const float* __restrict__ src,
                                bf16_t* __restrict__ dst, int n) {
    int i = blockIdx.x * 256 + threadIdx.x;
    if (i < n) dst[i] = f2bf(src[i]);
}

// Pack ConvTranspose2d weight (Cin, Cout, 4, 4), stride 2, pad 1 into
// 4 parity GEMM matrices: dst[(py*2+px)][Cout][Cin*4]
__global__ void pack_tconv_k4s2(const float* __restrict__ w,
                                bf16_t* __restrict__ dst, int Cin, int Cout) {
    int i = blockIdx.x * 256 + threadIdx.x;
    int total = 4 * Cout * Cin * 4;
    if (i >= total) return;
    int per = Cout * Cin * 4;
    int p  = i / per;
    int r  = i - p * per;
    int Kc = Cin * 4;
    int co = r / Kc;
    int q  = r - co * Kc;
    int ci = q >> 2;
    int tk = q & 3;
    int py = p >> 1, px = p & 1;
    int tky = tk >> 1, tkx = tk & 1;
    int ay = py + 2 * tky;
    int ax = px + 2 * tkx;
    float v = w[(((size_t)ci * Cout + co) * 4 + (3 - ay)) * 4 + (3 - ax)];
    dst[i] = f2bf(v);
}

__global__ void embed_norm2(const float* __restrict__ e, float* __restrict__ o) {
    int c = blockIdx.x * 64 + threadIdx.x;
    if (c >= 512) return;
    float s = 0.f;
    for (int j = 0; j < 128; ++j) { float v = e[c * 128 + j]; s += v * v; }
    o[c] = s;
}

__global__ void zero_f32(float* p, int n) {
    int i = blockIdx.x * 256 + threadIdx.x;
    if (i < n) p[i] = 0.f;
}

// VQ: argmin_c (||e_c||^2 - 2*dot(e_c, z_e)), gather z_q, accumulate VQ loss.
__global__ void vq_argmin_gather(const float* __restrict__ scores,
                                 const float* __restrict__ enorm,
                                 const float* __restrict__ embed,
                                 const float* __restrict__ ze_f32,
                                 float* __restrict__ idx_out,
                                 float* __restrict__ zq_out,
                                 bf16_t* __restrict__ zq_bf,
                                 float* __restrict__ loss_acc) {
    int n = blockIdx.x * 256 + threadIdx.x;      // 0 .. 32767
    const int NPIX = 8 * 64 * 64;
    if (n >= NPIX) return;
    int b   = n >> 12;
    int pix = n & 4095;
    const float* sc = scores + ((size_t)b * 512) * 4096 + pix;
    float best = 3.4e38f;
    int   bi   = 0;
    for (int c = 0; c < 512; ++c) {
        float dist = enorm[c] - 2.0f * sc[(size_t)c * 4096];
        if (dist < best) { best = dist; bi = c; }
    }
    idx_out[n] = (float)bi;
    float lsum = 0.f;
    for (int k = 0; k < 128; ++k) {
        float q = embed[(size_t)bi * 128 + k];
        size_t o = (((size_t)b * 128 + k) * 4096) + pix;  // (B,128,64,64)
        zq_out[o] = q;
        zq_bf[o]  = f2bf(q);
        float dd = q - ze_f32[o];
        lsum += dd * dd;
    }
    atomicAdd(loss_acc, lsum);
}

__global__ void finalize_loss(const float* __restrict__ acc, float* __restrict__ out) {
    out[0] = 0.25f * acc[0] / 4194304.0f;   // BETA * mean over (8,128,64,64)
}

// ---------------------------------------------------------------------------
// Host driver
// ---------------------------------------------------------------------------
extern "C" void kernel_launch(void* const* d_in, const int* in_sizes, int n_in,
                              void* d_out, int out_size, void* d_ws, size_t ws_size,
                              hipStream_t stream) {
    (void)in_sizes; (void)n_in; (void)out_size; (void)ws_size;

    const float* x     = (const float*)d_in[0];
    const float* e_w1  = (const float*)d_in[1];   const float* e_b1  = (const float*)d_in[2];
    const float* e_w2  = (const float*)d_in[3];   const float* e_b2  = (const float*)d_in[4];
    const float* e_w3  = (const float*)d_in[5];   const float* e_b3  = (const float*)d_in[6];
    const float* er1w1 = (const float*)d_in[7];   const float* er1b1 = (const float*)d_in[8];
    const float* er1w2 = (const float*)d_in[9];   const float* er1b2 = (const float*)d_in[10];
    const float* er2w1 = (const float*)d_in[11];  const float* er2b1 = (const float*)d_in[12];
    const float* er2w2 = (const float*)d_in[13];  const float* er2b2 = (const float*)d_in[14];
    const float* e_ow  = (const float*)d_in[15];  const float* e_ob  = (const float*)d_in[16];
    const float* embed = (const float*)d_in[17];
    const float* d_iw  = (const float*)d_in[18];  const float* d_ib  = (const float*)d_in[19];
    const float* dr1w1 = (const float*)d_in[20];  const float* dr1b1 = (const float*)d_in[21];
    const float* dr1w2 = (const float*)d_in[22];  const float* dr1b2 = (const float*)d_in[23];
    const float* dr2w1 = (const float*)d_in[24];  const float* dr2b1 = (const float*)d_in[25];
    const float* dr2w2 = (const float*)d_in[26];  const float* dr2b2 = (const float*)d_in[27];
    const float* d_tw1 = (const float*)d_in[28];  const float* d_tb1 = (const float*)d_in[29];
    const float* d_tw2 = (const float*)d_in[30];  const float* d_tb2 = (const float*)d_in[31];
    const float* d_ow  = (const float*)d_in[32];  const float* d_ob  = (const float*)d_in[33];

    float* out = (float*)d_out;
    const size_t N_XREC = (size_t)8 * 3 * 256 * 256;      // 1572864
    const size_t OFF_LOSS = N_XREC;
    const size_t OFF_IDX  = OFF_LOSS + 1;
    const size_t OFF_ZQ   = OFF_IDX + 32768;
    const size_t OFF_ZE   = OFF_ZQ + 4194304;

    // ---- workspace carve (regions reused across phases) ----
    char* ws = (char*)d_ws;
    size_t off = 0;
    auto take = [&](size_t bytes) -> void* {
        void* p = ws + off;
        off += (bytes + 255) & ~(size_t)255;
        return p;
    };
    bf16_t* xbf     = (bf16_t*)take((size_t)1572864 * 2);
    bf16_t* wb_e1   = (bf16_t*)take(6144   * 2);
    bf16_t* wb_e2   = (bf16_t*)take(524288 * 2);
    bf16_t* wb_e3   = (bf16_t*)take(294912 * 2);
    bf16_t* wb_er1a = (bf16_t*)take(147456 * 2);
    bf16_t* wb_er1b = (bf16_t*)take(16384  * 2);
    bf16_t* wb_er2a = (bf16_t*)take(147456 * 2);
    bf16_t* wb_er2b = (bf16_t*)take(16384  * 2);
    bf16_t* wb_eow  = (bf16_t*)take(16384  * 2);
    bf16_t* wb_emb  = (bf16_t*)take(65536  * 2);
    bf16_t* wb_diw  = (bf16_t*)take(16384  * 2);
    bf16_t* wb_dr1a = (bf16_t*)take(147456 * 2);
    bf16_t* wb_dr1b = (bf16_t*)take(16384  * 2);
    bf16_t* wb_dr2a = (bf16_t*)take(147456 * 2);
    bf16_t* wb_dr2b = (bf16_t*)take(16384  * 2);
    bf16_t* wb_t1   = (bf16_t*)take(524288 * 2);   // 4 parity mats [256][512]
    bf16_t* wb_t2   = (bf16_t*)take(524288 * 2);   // 4 parity mats [128][1024]
    bf16_t* wb_dow  = (bf16_t*)take(384    * 2);
    // big reusable regions
    void* R1 = take((size_t)67108864 * 2);  // h1 (33.6MB) -> later D2 (134MB)
    void* R2 = take((size_t)67108864);      // h2 (16.8MB) -> scores(67MB) -> D1(67MB)
    bf16_t* zA   = (bf16_t*)take((size_t)4194304 * 2);
    bf16_t* zB   = (bf16_t*)take((size_t)4194304 * 2);
    bf16_t* zT   = (bf16_t*)take((size_t)4194304 * 2);
    bf16_t* zebf = (bf16_t*)take((size_t)4194304 * 2);
    bf16_t* zqbf = (bf16_t*)take((size_t)4194304 * 2);
    float*  enorm = (float*)take(512 * 4);
    float*  lossA = (float*)take(256);

    bf16_t* h1     = (bf16_t*)R1;
    bf16_t* D2     = (bf16_t*)R1;
    bf16_t* h2     = (bf16_t*)R2;
    float*  scores = (float*)R2;
    bf16_t* D1     = (bf16_t*)R2;

    auto cvt = [&](const float* s, bf16_t* d, int n) {
        cvt_f32_to_bf16<<<(n + 255) / 256, 256, 0, stream>>>(s, d, n);
    };
    auto conv = [&](const bf16_t* X, const bf16_t* Wm, const float* bias,
                    const bf16_t* Res, bf16_t* Y, float* Yf,
                    int Bn, int Cin, int IH, int IW, int KH, int KW, int stride,
                    int pady, int padx, int OHc, int OWc, int OHf, int OWf,
                    int osy, int ooy, int osx, int oox, int M, int K,
                    int relu_in, int act) {
        ConvDesc dsc{X, Wm, bias, Res, Y, Yf, Bn, Cin, IH, IW, KH, KW, stride,
                     pady, padx, OHc, OWc, OHf, OWf, osy, ooy, osx, oox,
                     M, K, Bn * OHc * OWc, relu_in, act};
        dim3 grid((unsigned)((dsc.N + 63) / 64), (unsigned)((M + 127) / 128));
        conv_gemm_wmma<<<grid, 256, 0, stream>>>(dsc);
    };

    // ---- precision conversion / weight packing ----
    cvt(x, xbf, 1572864);
    cvt(e_w1, wb_e1, 6144);       cvt(e_w2, wb_e2, 524288);
    cvt(e_w3, wb_e3, 294912);
    cvt(er1w1, wb_er1a, 147456);  cvt(er1w2, wb_er1b, 16384);
    cvt(er2w1, wb_er2a, 147456);  cvt(er2w2, wb_er2b, 16384);
    cvt(e_ow, wb_eow, 16384);     cvt(embed, wb_emb, 65536);
    cvt(d_iw, wb_diw, 16384);
    cvt(dr1w1, wb_dr1a, 147456);  cvt(dr1w2, wb_dr1b, 16384);
    cvt(dr2w1, wb_dr2a, 147456);  cvt(dr2w2, wb_dr2b, 16384);
    cvt(d_ow, wb_dow, 384);
    pack_tconv_k4s2<<<(524288 + 255) / 256, 256, 0, stream>>>(d_tw1, wb_t1, 128, 256);
    pack_tconv_k4s2<<<(524288 + 255) / 256, 256, 0, stream>>>(d_tw2, wb_t2, 256, 128);
    embed_norm2<<<8, 64, 0, stream>>>(embed, enorm);
    zero_f32<<<1, 256, 0, stream>>>(lossA, 1);

    // ---- encoder ----
    conv(xbf, wb_e1, e_b1, nullptr, h1, nullptr,
         8, 3, 256, 256, 4, 4, 2, 1, 1, 128, 128, 128, 128, 1, 0, 1, 0, 128, 48, 0, 1);
    conv(h1, wb_e2, e_b2, nullptr, h2, nullptr,
         8, 128, 128, 128, 4, 4, 2, 1, 1, 64, 64, 64, 64, 1, 0, 1, 0, 256, 2048, 0, 1);
    conv(h2, wb_e3, e_b3, nullptr, zA, nullptr,
         8, 256, 64, 64, 3, 3, 1, 1, 1, 64, 64, 64, 64, 1, 0, 1, 0, 128, 2304, 0, 0);
    // resblock 1
    conv(zA, wb_er1a, er1b1, nullptr, zT, nullptr,
         8, 128, 64, 64, 3, 3, 1, 1, 1, 64, 64, 64, 64, 1, 0, 1, 0, 128, 1152, 1, 0);
    conv(zT, wb_er1b, er1b2, zA, zB, nullptr,
         8, 128, 64, 64, 1, 1, 1, 0, 0, 64, 64, 64, 64, 1, 0, 1, 0, 128, 128, 1, 0);
    // resblock 2
    conv(zB, wb_er2a, er2b1, nullptr, zT, nullptr,
         8, 128, 64, 64, 3, 3, 1, 1, 1, 64, 64, 64, 64, 1, 0, 1, 0, 128, 1152, 1, 0);
    conv(zT, wb_er2b, er2b2, zB, zA, nullptr,
         8, 128, 64, 64, 1, 1, 1, 0, 0, 64, 64, 64, 64, 1, 0, 1, 0, 128, 128, 1, 0);
    // z_e (bf16 for VQ GEMM + f32 to d_out)
    conv(zA, wb_eow, e_ob, nullptr, zebf, out + OFF_ZE,
         8, 128, 64, 64, 1, 1, 1, 0, 0, 64, 64, 64, 64, 1, 0, 1, 0, 128, 128, 0, 0);

    // ---- VQ: dot(e_c, z_e) via WMMA GEMM, then argmin/gather/loss ----
    conv(zebf, wb_emb, nullptr, nullptr, nullptr, scores,
         8, 128, 64, 64, 1, 1, 1, 0, 0, 64, 64, 64, 64, 1, 0, 1, 0, 512, 128, 0, 0);
    vq_argmin_gather<<<128, 256, 0, stream>>>(scores, enorm, embed, out + OFF_ZE,
                                              out + OFF_IDX, out + OFF_ZQ, zqbf, lossA);
    finalize_loss<<<1, 1, 0, stream>>>(lossA, out + OFF_LOSS);

    // ---- decoder ----
    conv(zqbf, wb_diw, d_ib, nullptr, zA, nullptr,
         8, 128, 64, 64, 1, 1, 1, 0, 0, 64, 64, 64, 64, 1, 0, 1, 0, 128, 128, 0, 0);
    conv(zA, wb_dr1a, dr1b1, nullptr, zT, nullptr,
         8, 128, 64, 64, 3, 3, 1, 1, 1, 64, 64, 64, 64, 1, 0, 1, 0, 128, 1152, 1, 0);
    conv(zT, wb_dr1b, dr1b2, zA, zB, nullptr,
         8, 128, 64, 64, 1, 1, 1, 0, 0, 64, 64, 64, 64, 1, 0, 1, 0, 128, 128, 1, 0);
    conv(zB, wb_dr2a, dr2b1, nullptr, zT, nullptr,
         8, 128, 64, 64, 3, 3, 1, 1, 1, 64, 64, 64, 64, 1, 0, 1, 0, 128, 1152, 1, 0);
    conv(zT, wb_dr2b, dr2b2, zB, zA, nullptr,
         8, 128, 64, 64, 1, 1, 1, 0, 0, 64, 64, 64, 64, 1, 0, 1, 0, 128, 128, 1, 0);

    // transposed conv 1: (8,128,64,64) -> relu -> (8,256,128,128), 4 parity passes
    for (int p = 0; p < 4; ++p) {
        int py = p >> 1, px = p & 1;
        conv(zA, wb_t1 + (size_t)p * 256 * 512, d_tb1, nullptr, D1, nullptr,
             8, 128, 64, 64, 2, 2, 1, 1 - py, 1 - px, 64, 64, 128, 128,
             2, py, 2, px, 256, 512, 1, 1);
    }
    // transposed conv 2: (8,256,128,128) -> (8,128,256,256), relu
    for (int p = 0; p < 4; ++p) {
        int py = p >> 1, px = p & 1;
        conv(D1, wb_t2 + (size_t)p * 128 * 1024, d_tb2, nullptr, D2, nullptr,
             8, 256, 128, 128, 2, 2, 1, 1 - py, 1 - px, 128, 128, 256, 256,
             2, py, 2, px, 128, 1024, 0, 1);
    }
    // final 1x1 conv + tanh -> x_rec (f32)
    conv(D2, wb_dow, d_ob, nullptr, nullptr, out,
         8, 128, 256, 256, 1, 1, 1, 0, 0, 256, 256, 256, 256, 1, 0, 1, 0, 3, 128, 0, 2);
}
